// CaptionNet_61443802136628
// MI455X (gfx1250) — matrix operand order repaired and checked by reading, not compiled, not run on hardware
//
#include <hip/hip_runtime.h>
#include <hip/hip_bf16.h>

// ---------------------------------------------------------------------------
// CaptionNet for MI455X (gfx1250, wave32, WMMA bf16)
// H=512 E=250 V=30522 B=32 NC=5 T=20 CNN=2048
// ---------------------------------------------------------------------------

#define Hh   512
#define Ee   250
#define Vv   30522
#define Bb   32
#define NCc  5
#define Tt   20
#define CNNc 2048

typedef __attribute__((ext_vector_type(16))) __bf16 v16bf;
typedef __attribute__((ext_vector_type(8)))  __bf16 bfx8;
typedef __attribute__((ext_vector_type(8)))  float  v8f;
typedef __attribute__((ext_vector_type(4)))  int    v4i_;

// ---- optional CDNA5 async global->LDS path (probe via __has_builtin) ------
#if defined(__has_builtin)
#if __has_builtin(__builtin_amdgcn_global_load_async_to_lds_b128) && \
    __has_builtin(__builtin_amdgcn_s_wait_asynccnt)
#define USE_ASYNC_LDS 1
#endif
#endif

#ifdef USE_ASYNC_LDS
// arg0: int4* in addrspace(1) (global src); arg1: LDS dst; imm offset; cpol.
__device__ __forceinline__ void async_copy16(const void* gsrc, void* ldst) {
  __builtin_amdgcn_global_load_async_to_lds_b128(
      (__attribute__((address_space(1))) v4i_*)(unsigned long long)gsrc,
      (__attribute__((address_space(3))) v4i_*)(unsigned int)(unsigned long long)ldst,
      0, 0);
}
#define ASYNC_COPY16(src, dst) async_copy16((const void*)(src), (void*)(dst))
#define ASYNC_WAIT() __builtin_amdgcn_s_wait_asynccnt(0)
#else
#define ASYNC_COPY16(src, dst) (*(bfx8*)(dst) = *(const bfx8*)(src))
#define ASYNC_WAIT()
#endif

// ---------------------------------------------------------------------------
__device__ __forceinline__ v8f wmma_bf16(v16bf a, v16bf b, v8f c) {
  // D = A(16x32) * B(32x16) + C, fp32 accumulate
  return __builtin_amdgcn_wmma_f32_16x16x32_bf16(false, a, false, b, (short)0, c,
                                                 false, false);
}

__device__ __forceinline__ v16bf ld2(const __bf16* p0, const __bf16* p1) {
  union { v16bf v; bfx8 h[2]; } u;
  u.h[0] = *(const bfx8*)p0;
  u.h[1] = *(const bfx8*)p1;
  return u.v;
}

__device__ __forceinline__ v16bf zero16() {
  v16bf v;
#pragma unroll
  for (int i = 0; i < 16; ++i) v[i] = (__bf16)0.0f;
  return v;
}

__device__ __forceinline__ float sigmoidf_(float x) {
  return 1.0f / (1.0f + __expf(-x));
}
__device__ __forceinline__ float tanhf_(float x) {
  if (x > 12.0f) return 1.0f;
  if (x < -12.0f) return -1.0f;
  float e = __expf(2.0f * x);
  return (e - 1.0f) / (e + 1.0f);
}

// ---------------------------------------------------------------------------
// Generic bf16 WMMA GEMM:  C[z][m][n] = A[m,:K] . B[n,:K] + bias[n]
// A: bf16 [.., lda] row-major    B: bf16 [N][ldb] (weights, row = output col)
// block = 256 threads (8 waves); block tile 32(M) x 256(N); K step 64,
// A tile double-buffered through LDS (async-to-LDS when available).
// grid = (M/32, ceil(N/256), zcount);  z: nc-part = z%zdiv, dir-part = z/zdiv
// ---------------------------------------------------------------------------
template <bool OUT_BF16, bool RELU, bool NBOUND>
__global__ void __launch_bounds__(256)
gemm_bf16(const __bf16* __restrict__ A, long lda, long sA,
          const __bf16* __restrict__ Bw, long ldb, long sB,
          const float* __restrict__ bias, long sBias,
          void* __restrict__ Cv, long ldc, long sC,
          int N, int K, int zdiv) {
  const int z  = blockIdx.z;
  const int zm = z % zdiv;
  const int zd = z / zdiv;
  const __bf16* Az = A + (size_t)zm * (size_t)sA;
  const __bf16* Bz = Bw + (size_t)zd * (size_t)sB;
  const float* biasz = bias ? (bias + (size_t)zd * (size_t)sBias) : nullptr;
  char* Cz = (char*)Cv + (size_t)z * (size_t)sC * (OUT_BF16 ? 2u : 4u);

  const int m0   = blockIdx.x * 32;
  const int nblk = blockIdx.y * 256;
  const int tid  = threadIdx.x;
  const int lane = tid & 31;
  const int wave = tid >> 5;
  const int ln   = lane & 15;  // N (B/C) or M (A) index within tile
  const int lh   = lane >> 4;  // half selector
  const int n0w  = nblk + wave * 32;

  __shared__ __bf16 At[2][32][72];  // 32 x 64 tile, row padded to 72 (144B)

  // A-tile copy mapping: 256 threads x 16B = 32 rows x 64 bf16
  const int crow   = tid >> 3;
  const int cchunk = (tid & 7) * 8;

  const int nks = K >> 6;  // K must be a multiple of 64

  // preload stage 0
  {
    const __bf16* src = Az + (size_t)(m0 + crow) * lda + cchunk;
    ASYNC_COPY16(src, &At[0][crow][cchunk]);
  }

  v8f acc[2][2];
#pragma unroll
  for (int mt = 0; mt < 2; ++mt)
#pragma unroll
    for (int nt = 0; nt < 2; ++nt)
#pragma unroll
      for (int r = 0; r < 8; ++r) acc[mt][nt][r] = 0.0f;

  for (int ks = 0; ks < nks; ++ks) {
    ASYNC_WAIT();
    __syncthreads();
    if (ks + 1 < nks) {
      const __bf16* src = Az + (size_t)(m0 + crow) * lda + (ks + 1) * 64 + cchunk;
      ASYNC_COPY16(src, &At[(ks + 1) & 1][crow][cchunk]);
    }
    const int buf = ks & 1;
#pragma unroll
    for (int kbi = 0; kbi < 2; ++kbi) {
      const int kof = kbi * 32 + lh * 8;
      v16bf a0 = ld2(&At[buf][ln][kof],      &At[buf][ln][kof + 16]);
      v16bf a1 = ld2(&At[buf][16 + ln][kof], &At[buf][16 + ln][kof + 16]);
#pragma unroll
      for (int nt = 0; nt < 2; ++nt) {
        const int n = n0w + nt * 16 + ln;
        v16bf bf;
        const __bf16* brow = Bz + (size_t)n * ldb + ks * 64 + kbi * 32 + lh * 8;
        if (!NBOUND || n < N) {
          bf = ld2(brow, brow + 16);
          __builtin_prefetch(brow + 64, 0, 3);  // global_prefetch_b8, next stage
        } else {
          bf = zero16();
        }
        acc[0][nt] = wmma_bf16(a0, bf, acc[0][nt]);
        acc[1][nt] = wmma_bf16(a1, bf, acc[1][nt]);
      }
    }
  }

  // store C (C layout: lane n = ln, rows lh*8 + r)
#pragma unroll
  for (int mt = 0; mt < 2; ++mt) {
#pragma unroll
    for (int nt = 0; nt < 2; ++nt) {
      const int n = n0w + nt * 16 + ln;
      if (NBOUND && n >= N) continue;
      const float bv = biasz ? biasz[n] : 0.0f;
#pragma unroll
      for (int r = 0; r < 8; ++r) {
        const int m = m0 + mt * 16 + lh * 8 + r;
        float v = acc[mt][nt][r] + bv;
        if (RELU) v = fmaxf(v, 0.0f);
        if (OUT_BF16)
          ((__bf16*)Cz)[(size_t)m * ldc + n] = (__bf16)v;
        else
          ((float*)Cz)[(size_t)m * ldc + n] = v;
      }
    }
  }
}

// ---------------------------------------------------------------------------
// LSTM recurrence over one prefix.
// grid = (2 dirs, NC captions); block = 1024 threads (32 waves).
// Each wave owns h-columns [wave*16, wave*16+16) and computes the 4 matched
// gate tiles (i/f/g/o) with WMMA against LDS-resident h; c stays in regs.
// xproj holds the precomputed x@Wih^T + (bih+bhh) for every t.
// IS_L0: write per-step h into y (concat fwd|bwd halves);
// else : write final h of the prefix into packed [bwd | fwd].
// ---------------------------------------------------------------------------
template <bool IS_L0>
__global__ void __launch_bounds__(1024)
lstm_rec(const float* __restrict__ xproj, long sXdir, long sXnc,
         const __bf16* __restrict__ Whh,  // [2][2048][512]
         __bf16* __restrict__ h_state, float* __restrict__ c_state,
         int layerbase, __bf16* __restrict__ yout, int L, int w) {
  const int dir = blockIdx.x;
  const int nc  = blockIdx.y;
  const float* xp0 = xproj + (size_t)dir * sXdir + (size_t)nc * sXnc;
  const __bf16* Wz = Whh + (size_t)dir * 2048 * 512;
  const int sidx = nc * 4 + layerbase + dir;
  __bf16* hst = h_state + (size_t)sidx * Bb * Hh;
  float*  cst = c_state + (size_t)sidx * Bb * Hh;

  const int tid  = threadIdx.x;
  const int lane = tid & 31;
  const int wave = tid >> 5;
  const int ln   = lane & 15;
  const int lh   = lane >> 4;
  const int nn   = wave * 16 + ln;  // this lane's h column

  __shared__ __bf16 hsh[32][520];  // 32 x 512 h, row padded (1040B = 65*16)

  float creg[2][8];
#pragma unroll
  for (int mt = 0; mt < 2; ++mt)
#pragma unroll
    for (int r = 0; r < 8; ++r) {
      const int m = mt * 16 + lh * 8 + r;
      creg[mt][r] = cst[(size_t)m * Hh + nn];
      hsh[m][nn]  = hst[(size_t)m * Hh + nn];
    }
  __syncthreads();

  for (int s = 0; s < L; ++s) {
    const int t = (dir == 0) ? s : (L - 1 - s);
    const float* xp = xp0 + (size_t)t * (Bb * 2048);

    v8f acc[2][4];
#pragma unroll
    for (int mt = 0; mt < 2; ++mt)
#pragma unroll
      for (int g = 0; g < 4; ++g)
#pragma unroll
        for (int r = 0; r < 8; ++r) {
          const int m = mt * 16 + lh * 8 + r;
          acc[mt][g][r] = xp[(size_t)m * 2048 + g * Hh + nn];
        }

#pragma unroll 4
    for (int kb = 0; kb < Hh / 32; ++kb) {
      const int kof = kb * 32 + lh * 8;
      v16bf a0 = ld2(&hsh[ln][kof],      &hsh[ln][kof + 16]);
      v16bf a1 = ld2(&hsh[16 + ln][kof], &hsh[16 + ln][kof + 16]);
#pragma unroll
      for (int g = 0; g < 4; ++g) {
        const __bf16* brow = Wz + (size_t)(g * Hh + nn) * Hh + kb * 32 + lh * 8;
        v16bf bf = ld2(brow, brow + 16);
        __builtin_prefetch(brow + 32, 0, 3);
        acc[0][g] = wmma_bf16(a0, bf, acc[0][g]);
        acc[1][g] = wmma_bf16(a1, bf, acc[1][g]);
      }
    }
    __syncthreads();  // all waves done reading hsh for this step

#pragma unroll
    for (int mt = 0; mt < 2; ++mt)
#pragma unroll
      for (int r = 0; r < 8; ++r) {
        const float ig = sigmoidf_(acc[mt][0][r]);
        const float fg = sigmoidf_(acc[mt][1][r]);
        const float gg = tanhf_(acc[mt][2][r]);
        const float og = sigmoidf_(acc[mt][3][r]);
        const float c  = fg * creg[mt][r] + ig * gg;
        creg[mt][r]    = c;
        const float h  = og * tanhf_(c);
        const __bf16 hb = (__bf16)h;
        const int m = mt * 16 + lh * 8 + r;
        hsh[m][nn] = hb;
        if (IS_L0)
          yout[((size_t)(nc * Tt + t) * Bb + m) * (2 * Hh) + dir * Hh + nn] = hb;
        if (s == L - 1) {
          if (!IS_L0)  // packed = [layer1-bwd | layer1-fwd]
            yout[((size_t)(nc * Tt + w) * Bb + m) * (2 * Hh) + (dir ? 0 : Hh) + nn] = hb;
          hst[(size_t)m * Hh + nn] = hb;
          cst[(size_t)m * Hh + nn] = c;
        }
      }
    __syncthreads();  // new h visible before next step's reads
  }
}

// ---------------------------------------------------------------------------
// small helper kernels
// ---------------------------------------------------------------------------
__global__ void cvt_pad_kernel(const float* __restrict__ src, __bf16* __restrict__ dst,
                               int K, int Kp, long total) {
  long i = (long)blockIdx.x * blockDim.x + threadIdx.x;
  if (i >= total) return;
  long row = i / Kp;
  int  col = (int)(i % Kp);
  float v = (col < K) ? src[row * K + col] : 0.0f;
  dst[i] = (__bf16)v;
}

__global__ void bias_sum_kernel(const float* a, const float* b, float* o, int n) {
  int i = blockIdx.x * blockDim.x + threadIdx.x;
  if (i < n) o[i] = a[i] + b[i];
}

__global__ void embed_kernel(const int* __restrict__ ix, const float* __restrict__ emb,
                             __bf16* __restrict__ xemb) {
  long i = (long)blockIdx.x * blockDim.x + threadIdx.x;  // over 3200*256
  if (i >= (long)NCc * Tt * Bb * 256) return;
  int  col = (int)(i & 255);
  long row = i >> 8;                       // (nc*T + t)*B + b
  int b  = (int)(row % Bb);
  long r2 = row / Bb;
  int t  = (int)(r2 % Tt);
  int nc = (int)(r2 / Tt);
  int tok = ix[(b * NCc + nc) * Tt + t];
  float v = (col < Ee) ? emb[(size_t)tok * Ee + col] : 0.0f;
  xemb[i] = (__bf16)v;
}

__global__ void replicate_state_kernel(const __bf16* __restrict__ h0,
                                       const float* __restrict__ c0,
                                       __bf16* __restrict__ hstate,
                                       float* __restrict__ cstate) {
  long i = (long)blockIdx.x * blockDim.x + threadIdx.x;  // over 20*16384
  if (i >= (long)NCc * 4 * Bb * Hh) return;
  long j = i % (Bb * Hh);
  hstate[i] = h0[j];
  cstate[i] = c0[j];
}

// ---------------------------------------------------------------------------
extern "C" void kernel_launch(void* const* d_in, const int* in_sizes, int n_in,
                              void* d_out, int out_size, void* d_ws, size_t ws_size,
                              hipStream_t stream) {
  (void)in_sizes; (void)n_in; (void)out_size; (void)ws_size;

  const float* img   = (const float*)d_in[0];
  const int*   caps  = (const int*)d_in[1];
  const float* emb   = (const float*)d_in[2];
  const float* Wh    = (const float*)d_in[3];
  const float* bh    = (const float*)d_in[4];
  const float* Wc    = (const float*)d_in[5];
  const float* bc    = (const float*)d_in[6];
  const float* Wih0  = (const float*)d_in[7];
  const float* Whh0  = (const float*)d_in[8];
  const float* bih0  = (const float*)d_in[9];
  const float* bhh0  = (const float*)d_in[10];
  const float* Wih1  = (const float*)d_in[11];
  const float* Whh1  = (const float*)d_in[12];
  const float* bih1  = (const float*)d_in[13];
  const float* bhh1  = (const float*)d_in[14];
  const float* Wfc   = (const float*)d_in[15];
  const float* bfc   = (const float*)d_in[16];

  // bump allocator over d_ws (total ~207 MB)
  size_t off = 0;
  auto alloc = [&](size_t bytes) -> char* {
    char* p = (char*)d_ws + off;
    off = (off + bytes + 255) & ~(size_t)255;
    return p;
  };
  __bf16* img_bf = (__bf16*)alloc((size_t)Bb * CNNc * 2);
  __bf16* whc    = (__bf16*)alloc((size_t)2 * Hh * CNNc * 2);       // Wh | Wc
  __bf16* wih0   = (__bf16*)alloc((size_t)2 * 2048 * 256 * 2);      // K padded 250->256
  __bf16* whh0   = (__bf16*)alloc((size_t)2 * 2048 * 512 * 2);
  __bf16* wih1   = (__bf16*)alloc((size_t)2 * 2048 * 1024 * 2);
  __bf16* whh1   = (__bf16*)alloc((size_t)2 * 2048 * 512 * 2);
  __bf16* wfc    = (__bf16*)alloc((size_t)Vv * 1024 * 2);
  float*  b0     = (float*)alloc((size_t)2 * 2048 * 4);
  float*  b1     = (float*)alloc((size_t)2 * 2048 * 4);
  __bf16* xemb   = (__bf16*)alloc((size_t)NCc * Tt * Bb * 256 * 2);
  float*  xproj0 = (float*)alloc((size_t)2 * NCc * Tt * Bb * 2048 * 4);
  float*  xproj1 = (float*)alloc((size_t)2 * NCc * Tt * Bb * 2048 * 4);
  __bf16* ybuf   = (__bf16*)alloc((size_t)NCc * Tt * Bb * 1024 * 2);
  __bf16* packed = (__bf16*)alloc((size_t)NCc * Tt * Bb * 1024 * 2);
  __bf16* hstate = (__bf16*)alloc((size_t)NCc * 4 * Bb * Hh * 2);
  float*  cstate = (float*)alloc((size_t)NCc * 4 * Bb * Hh * 4);
  __bf16* h0tmp  = (__bf16*)alloc((size_t)Bb * Hh * 2);
  float*  c0tmp  = (float*)alloc((size_t)Bb * Hh * 4);

  auto cvt = [&](const float* s, __bf16* d, int K, int Kp, long rows) {
    long total = rows * (long)Kp;
    cvt_pad_kernel<<<dim3((unsigned)((total + 255) / 256)), 256, 0, stream>>>(s, d, K, Kp, total);
  };

  // 1) weight / activation conversions to bf16 (padded where needed)
  cvt(img,  img_bf, CNNc, CNNc, Bb);
  cvt(Wh,   whc,                CNNc, CNNc, Hh);
  cvt(Wc,   whc + (size_t)Hh * CNNc, CNNc, CNNc, Hh);
  cvt(Wih0, wih0, Ee,   256,  2 * 2048);
  cvt(Whh0, whh0, 512,  512,  2 * 2048);
  cvt(Wih1, wih1, 1024, 1024, 2 * 2048);
  cvt(Whh1, whh1, 512,  512,  2 * 2048);
  cvt(Wfc,  wfc,  1024, 1024, Vv);

  // 2) fused gate biases
  bias_sum_kernel<<<dim3(16), 256, 0, stream>>>(bih0, bhh0, b0, 2 * 2048);
  bias_sum_kernel<<<dim3(16), 256, 0, stream>>>(bih1, bhh1, b1, 2 * 2048);

  // 3) embedding gather (padded K=256)
  embed_kernel<<<dim3((NCc * Tt * Bb * 256) / 256), 256, 0, stream>>>(caps, emb, xemb);

  // 4) h0 = relu(img@Wh^T + bh) [bf16], c0 = relu(img@Wc^T + bc) [f32]
  gemm_bf16<true, true, false><<<dim3(1, 2, 1), 256, 0, stream>>>(
      img_bf, CNNc, 0, whc, CNNc, 0, bh, 0, h0tmp, Hh, 0, Hh, CNNc, 1);
  gemm_bf16<false, true, false><<<dim3(1, 2, 1), 256, 0, stream>>>(
      img_bf, CNNc, 0, whc + (size_t)Hh * CNNc, CNNc, 0, bc, 0, c0tmp, Hh, 0, Hh, CNNc, 1);
  replicate_state_kernel<<<dim3((NCc * 4 * Bb * Hh) / 256), 256, 0, stream>>>(
      h0tmp, c0tmp, hstate, cstate);

  // 5) layer-0 input projections for ALL timesteps, once:
  //    xproj0[dir][nc,t,b][2048] = xemb @ Wih0[dir]^T + b0[dir]
  gemm_bf16<false, false, false><<<dim3(100, 8, 2), 256, 0, stream>>>(
      xemb, 256, 0, wih0, 256, (long)2048 * 256, b0, 2048,
      xproj0, 2048, (long)NCc * Tt * Bb * 2048, 2048, 256, 1);

  // 6) prefix loop (faithful O(T^2) with carried states)
  for (int w = 0; w < Tt; ++w) {
    const int L = w + 1;
    // layer 0 recurrence -> y[nc][t][b][fwd|bwd]
    lstm_rec<true><<<dim3(2, NCc), 1024, 0, stream>>>(
        xproj0, (long)NCc * Tt * Bb * 2048, (long)Tt * Bb * 2048,
        whh0, hstate, cstate, 0, ybuf, L, w);
    // layer-1 input projections for this prefix (parallel over t):
    // xproj1[dir][nc][t][b][2048] = y @ Wih1[dir]^T + b1[dir]
    gemm_bf16<false, false, false><<<dim3(L, 8, 2 * NCc), 256, 0, stream>>>(
        ybuf, 1024, (long)Tt * Bb * 1024, wih1, 1024, (long)2048 * 1024,
        b1, 2048, xproj1, 2048, (long)Tt * Bb * 2048, 2048, 1024, NCc);
    // layer 1 recurrence -> packed[nc][w][b][l1bwd|l1fwd]
    lstm_rec<false><<<dim3(2, NCc), 1024, 0, stream>>>(
        xproj1, (long)NCc * Tt * Bb * 2048, (long)Tt * Bb * 2048,
        whh1, hstate, cstate, 2, packed, L, w);
  }

  // 7) final FC: out[nc,t,b][V] = packed @ Wfc^T + bfc  (fp32 out, ragged N)
  gemm_bf16<false, false, true><<<dim3(100, 120, 1), 256, 0, stream>>>(
      packed, 1024, 0, wfc, 1024, 0, bfc, 0, (float*)d_out, Vv, 0, Vv, 1024, 1);
}
